// SpikeDrivenTransformer_79216376807575
// MI455X (gfx1250) — matrix operand order repaired and checked by reading, not compiled, not run on hardware
//
#include <hip/hip_runtime.h>
#include <hip/hip_bf16.h>

typedef __attribute__((ext_vector_type(16))) _Float16 v16h;
typedef __attribute__((ext_vector_type(8)))  _Float16 v8h;
typedef __attribute__((ext_vector_type(8)))  float    v8f;

#define TT   4
#define BB   16
#define VV   512
#define EE   512
#define HID  2048
#define NN   (TT*BB*VV)   /* 32768 columns */
#define BV   (BB*VV)      /* 8192 */
#define NCLS 11
#define BN_SCALE 0.9999950000374996f   /* 1/sqrt(1+1e-5) */

// ---------------------------------------------------------------------------
// fp32 -> fp16 weight conversion (weights only; spikes are exact in fp16)
// ---------------------------------------------------------------------------
__global__ void k_f32_to_f16(const float* __restrict__ s, _Float16* __restrict__ d, long n) {
  long i = (long)blockIdx.x * 256 + threadIdx.x;
  if (i < n) d[i] = (_Float16)s[i];
}

// ---------------------------------------------------------------------------
// Patch embed: y[n,o] = (w[o,0]*x0 + w[o,1]*x1) * g[o]*BN_SCALE + b[o]
// ---------------------------------------------------------------------------
__global__ void k_embed(const float* __restrict__ x, const float* __restrict__ w,
                        const float* __restrict__ g, const float* __restrict__ b,
                        float* __restrict__ y) {
  int n  = blockIdx.x;
  int o  = threadIdx.x + blockIdx.y * blockDim.x;
  int tb = n / VV, v = n % VV;
  float x0 = x[(tb * 2 + 0) * VV + v];
  float x1 = x[(tb * 2 + 1) * VV + v];
  float acc = w[o * 2 + 0] * x0 + w[o * 2 + 1] * x1;
  acc = acc * (g[o] * BN_SCALE) + b[o];
  y[(long)n * EE + o] = acc;
}

// ---------------------------------------------------------------------------
// Multi-step LIF (tau=2, hard reset).  Layout y[t*states + i].
// ---------------------------------------------------------------------------
template <typename OutT>
__global__ void k_lif(const float* __restrict__ y, OutT* __restrict__ s,
                      long states, float vth) {
  long i = (long)blockIdx.x * 256 + threadIdx.x;
  if (i >= states) return;
  float vm = 0.f;
#pragma unroll
  for (int t = 0; t < TT; ++t) {
    float xv = y[(long)t * states + i];
    vm += (xv - vm) * 0.5f;
    float sp = (vm >= vth) ? 1.f : 0.f;
    s[(long)t * states + i] = (OutT)sp;
    vm *= (1.f - sp);
  }
}

// ---------------------------------------------------------------------------
// one 16x16x32 WMMA against a streamed B sub-tile (two contiguous b128 loads)
// ---------------------------------------------------------------------------
__device__ __forceinline__ v8f wmma_step(const _Float16* __restrict__ bptr, int ko,
                                         v16h av, v8f acc) {
  v8h blo = *(const v8h*)(bptr + ko);
  v8h bhi = *(const v8h*)(bptr + ko + 8);
  v16h bv = __builtin_shufflevector(blo, bhi, 0,1,2,3,4,5,6,7,8,9,10,11,12,13,14,15);
  return __builtin_amdgcn_wmma_f32_16x16x32_f16(false, av, false, bv,
                                                (short)0, acc, false, false);
}

// ---------------------------------------------------------------------------
// WMMA GEMM:  Y[n,m] = sum_k A[m,k]*Bt[n,k]  (+convb)*g*BN_SCALE+bnb (+res)
// A: fp16 [M,K] weights, Bt: fp16 [N,K] spikes, Y: fp32 [N,M].
// Block = 256 thr = 8 waves; block tile 32(M) x 256(N); wave tile 16 x 64
// (4 accumulators -> 4 WMMA per A fragment).  A is staged in LDS in 64-wide
// K chunks, double-buffered via CDNA5 global_load_async_to_lds_b128 and
// retired with s_wait_asynccnt; one barrier per chunk (8 WMMA / barrier).
// ---------------------------------------------------------------------------
__global__ __launch_bounds__(256)
void k_wgemm(const _Float16* __restrict__ A, const _Float16* __restrict__ Bt,
             float* __restrict__ Y,
             const float* __restrict__ convb, const float* __restrict__ g,
             const float* __restrict__ bnb, const float* __restrict__ res,
             int M, int K) {
  __shared__ _Float16 sA[2][32][64];   // 2 x 4KB double buffer

  const int m0 = blockIdx.y * 32;
  const int n0 = blockIdx.x * 256;
  const int tid  = threadIdx.x;
  const int lane = tid & 31, wave = tid >> 5;
  const int wm = wave >> 2, wn = wave & 3;       // 2(M) x 4(N) wave grid
  const int hi = lane >> 4, ln = lane & 15;

  // staging: one b128 (8 halves) per thread covers the 32x64 chunk
  const int arow = tid >> 3;
  const int acol = (tid & 7) << 3;
  const _Float16* aSrc = A + (size_t)(m0 + arow) * K + acol;

  // B pointers for this wave's 4 n-subtiles (16 contiguous K halves per lane)
  const _Float16* bp0 = Bt + (size_t)(n0 + wn * 64 +  0 + ln) * K + hi * 16;
  const _Float16* bp1 = Bt + (size_t)(n0 + wn * 64 + 16 + ln) * K + hi * 16;
  const _Float16* bp2 = Bt + (size_t)(n0 + wn * 64 + 32 + ln) * K + hi * 16;
  const _Float16* bp3 = Bt + (size_t)(n0 + wn * 64 + 48 + ln) * K + hi * 16;

  v8f acc0 = {}, acc1 = {}, acc2 = {}, acc3 = {};

  // prologue: async-stage chunk 0 into buffer 0
  {
    unsigned ldo = (unsigned)(uintptr_t)&sA[0][arow][acol];
    unsigned long long ga = (unsigned long long)(uintptr_t)aSrc;
    asm volatile("global_load_async_to_lds_b128 %0, %1, off"
                 :: "v"(ldo), "v"(ga) : "memory");
  }
  asm volatile("s_wait_asynccnt 0x0" ::: "memory");
  __syncthreads();

  const int C = K >> 6;
  int buf = 0;
  for (int c = 0; c < C; ++c) {
    // issue async stage of next chunk into the other buffer
    if (c + 1 < C) {
      unsigned ldo = (unsigned)(uintptr_t)&sA[buf ^ 1][arow][acol];
      unsigned long long ga = (unsigned long long)(uintptr_t)(aSrc + (size_t)(c + 1) * 64);
      asm volatile("global_load_async_to_lds_b128 %0, %1, off"
                   :: "v"(ldo), "v"(ga) : "memory");
    }
    const _Float16* aL = &sA[buf][wm * 16 + ln][hi * 16];
    const int kb = c << 6;
#pragma unroll
    for (int s = 0; s < 2; ++s) {
      // A fragment from LDS: contiguous run [k0..k0+15], k0 = 16*hi
      v8h alo = *(const v8h*)(aL + s * 32);        // ds_load_b128
      v8h ahi = *(const v8h*)(aL + s * 32 + 8);    // ds_load_b128
      v16h av = __builtin_shufflevector(alo, ahi, 0,1,2,3,4,5,6,7,8,9,10,11,12,13,14,15);
      const int ko = kb + s * 32;
      acc0 = wmma_step(bp0, ko, av, acc0);
      acc1 = wmma_step(bp1, ko, av, acc1);
      acc2 = wmma_step(bp2, ko, av, acc2);
      acc3 = wmma_step(bp3, ko, av, acc3);
    }
    __builtin_prefetch(bp0 + kb + 512, 0, 1);      // global_prefetch_b8
    asm volatile("s_wait_asynccnt 0x0" ::: "memory");
    __syncthreads();
    buf ^= 1;
  }

  // epilogue: per lane, per subtile: 8 contiguous floats (M = mstart..+7)
  const int mstart = m0 + wm * 16 + hi * 8;
#pragma unroll
  for (int j = 0; j < 4; ++j) {
    v8f a = (j == 0) ? acc0 : (j == 1) ? acc1 : (j == 2) ? acc2 : acc3;
    const int n = n0 + wn * 64 + j * 16 + ln;
    float* yp = Y + (size_t)n * M + mstart;
    const float* rp = res ? res + (size_t)n * M + mstart : nullptr;
#pragma unroll
    for (int r = 0; r < 8; ++r) {
      int m = mstart + r;
      float val = a[r];
      if (convb) val += convb[m];
      val = val * (g[m] * BN_SCALE) + bnb[m];
      if (rp) val += rp[r];
      yp[r] = val;
    }
  }
}

// ---------------------------------------------------------------------------
// kv[tb,c] = sum_v k[(tb,v),c] * v[(tb,v),c]
// ---------------------------------------------------------------------------
__global__ void k_kv_reduce(const _Float16* __restrict__ k, const _Float16* __restrict__ v,
                            float* __restrict__ out) {
  int tb = blockIdx.x;
  int c  = threadIdx.x + blockIdx.y * blockDim.x;
  const _Float16* kp = k + (size_t)tb * VV * EE + c;
  const _Float16* vp = v + (size_t)tb * VV * EE + c;
  float s = 0.f;
  for (int j = 0; j < VV; ++j) s += (float)kp[(size_t)j * EE] * (float)vp[(size_t)j * EE];
  out[(size_t)tb * EE + c] = s;
}

__global__ void k_mean_v(const float* __restrict__ h, float* __restrict__ out) {
  int tb = blockIdx.x;
  int c  = threadIdx.x + blockIdx.y * blockDim.x;
  const float* p = h + (size_t)tb * VV * EE + c;
  float s = 0.f;
  for (int j = 0; j < VV; ++j) s += p[(size_t)j * EE];
  out[(size_t)tb * EE + c] = s * (1.f / VV);
}

__global__ void k_qkv_mul(const _Float16* __restrict__ q, const _Float16* __restrict__ kvs,
                          _Float16* __restrict__ o) {
  long i = (long)blockIdx.x * 256 + threadIdx.x;
  if (i >= (long)NN * EE) return;
  long n = i >> 9;
  int  c = (int)(i & (EE - 1));
  int  tb = (int)(n >> 9);
  o[i] = q[i] * kvs[(size_t)tb * EE + c];
}

__global__ void k_head(const _Float16* __restrict__ hs, const float* __restrict__ w,
                       const float* __restrict__ hb, float* __restrict__ out) {
  int b = blockIdx.x;
  int o = threadIdx.x;
  if (o >= NCLS) return;
  float s = 0.f;
  for (int t = 0; t < TT; ++t) {
    const _Float16* hp = hs + (size_t)(t * BB + b) * EE;
    float d = 0.f;
    for (int c = 0; c < EE; ++c) d += w[o * EE + c] * (float)hp[c];
    s += d;
  }
  out[b * NCLS + o] = s * (1.f / TT) + hb[o];
}

// ---------------------------------------------------------------------------
extern "C" void kernel_launch(void* const* d_in, const int* in_sizes, int n_in,
                              void* d_out, int out_size, void* d_ws, size_t ws_size,
                              hipStream_t stream) {
  (void)in_sizes; (void)n_in; (void)out_size; (void)ws_size;
  const float* x  = (const float*)d_in[0];
  const float* ew = (const float*)d_in[1];
  const float* eg = (const float*)d_in[2];
  const float* eb = (const float*)d_in[3];
  const float* hw = (const float*)d_in[4];
  const float* hb = (const float*)d_in[5];

  char* ws = (char*)d_ws; size_t off = 0;
  auto carve = [&](size_t bytes) -> void* {
    void* p = ws + off; off += (bytes + 255) & ~(size_t)255; return p;
  };
  float*    h   = (float*)   carve((size_t)NN * EE * 4);
  float*    pre = (float*)   carve((size_t)NN * HID * 4);
  _Float16* sq  = (_Float16*)carve((size_t)NN * EE * 2);
  _Float16* sk  = (_Float16*)carve((size_t)NN * EE * 2);
  _Float16* sv  = (_Float16*)carve((size_t)NN * EE * 2);
  _Float16* sb  = (_Float16*)carve((size_t)NN * HID * 2);
  float*    kvp = (float*)   carve((size_t)TT * BB * EE * 4);
  _Float16* kvs = (_Float16*)carve((size_t)TT * BB * EE * 2);
  float*    hm  = (float*)   carve((size_t)TT * BB * EE * 4);
  _Float16* hms = (_Float16*)carve((size_t)TT * BB * EE * 2);
  _Float16* wt  = (_Float16*)carve((size_t)HID * EE * 2);

  dim3 b256(256);
  const long SE = (long)BV * EE;
  const long SH = (long)BV * HID;
  const long SB = (long)BB * EE;

  auto conv16 = [&](const float* src, long n) {
    k_f32_to_f16<<<(unsigned)((n + 255) / 256), b256, 0, stream>>>(src, wt, n);
  };
  auto gemm = [&](const _Float16* Bmat, float* Yo, const float* cb, const float* gg,
                  const float* bbb, const float* rr, int M, int K) {
    k_wgemm<<<dim3(NN / 256, M / 32), b256, 0, stream>>>(wt, Bmat, Yo, cb, gg, bbb, rr, M, K);
  };

  // patch embed -> LIF -> h (float spikes)
  k_embed<<<dim3(NN, EE / 256), b256, 0, stream>>>(x, ew, eg, eb, pre);
  k_lif<float><<<(unsigned)((SE + 255) / 256), b256, 0, stream>>>(pre, h, SE, 1.0f);

  for (int bl = 0; bl < 6; ++bl) {
    const float* P[21];
    for (int j = 0; j < 21; ++j) P[j] = (const float*)d_in[6 + 21 * bl + j];

    // xs = lif(h)
    k_lif<_Float16><<<(unsigned)((SE + 255) / 256), b256, 0, stream>>>(h, sb, SE, 1.0f);
    // q / k / v
    conv16(P[0], (long)EE * EE);
    gemm(sb, pre, nullptr, P[1], P[2], nullptr, EE, EE);
    k_lif<_Float16><<<(unsigned)((SE + 255) / 256), b256, 0, stream>>>(pre, sq, SE, 1.0f);
    conv16(P[3], (long)EE * EE);
    gemm(sb, pre, nullptr, P[4], P[5], nullptr, EE, EE);
    k_lif<_Float16><<<(unsigned)((SE + 255) / 256), b256, 0, stream>>>(pre, sk, SE, 1.0f);
    conv16(P[6], (long)EE * EE);
    gemm(sb, pre, nullptr, P[7], P[8], nullptr, EE, EE);
    k_lif<_Float16><<<(unsigned)((SE + 255) / 256), b256, 0, stream>>>(pre, sv, SE, 1.0f);
    // kv reduce + talking-heads LIF (vth = 0.5)
    k_kv_reduce<<<dim3(TT * BB, EE / 256), b256, 0, stream>>>(sk, sv, kvp);
    k_lif<_Float16><<<(unsigned)((SB + 255) / 256), b256, 0, stream>>>(kvp, kvs, SB, 0.5f);
    // o = q * kv
    k_qkv_mul<<<(unsigned)(((long)NN * EE + 255) / 256), b256, 0, stream>>>(sq, kvs, sb);
    // proj (+bias+BN+residual, in place into h)
    conv16(P[9], (long)EE * EE);
    gemm(sb, h, P[10], P[11], P[12], h, EE, EE);
    // MLP
    k_lif<_Float16><<<(unsigned)((SE + 255) / 256), b256, 0, stream>>>(h, sq, SE, 1.0f);
    conv16(P[13], (long)HID * EE);
    gemm(sq, pre, P[14], P[15], P[16], nullptr, HID, EE);
    k_lif<_Float16><<<(unsigned)((SH + 255) / 256), b256, 0, stream>>>(pre, sb, SH, 1.0f);
    conv16(P[17], (long)EE * HID);
    gemm(sb, h, P[18], P[19], P[20], h, EE, HID);
  }

  k_mean_v<<<dim3(TT * BB, EE / 256), b256, 0, stream>>>(h, hm);
  k_lif<_Float16><<<(unsigned)((SB + 255) / 256), b256, 0, stream>>>(hm, hms, SB, 1.0f);
  k_head<<<BB, 16, 0, stream>>>(hms, hw, hb, (float*)d_out);
}
